// GlobalPointer_1022202216945
// MI455X (gfx1250) — matrix operand order repaired and checked by reading, not compiled, not run on hardware
//
#include <hip/hip_runtime.h>
#include <math.h>

typedef __attribute__((ext_vector_type(16))) _Float16 v16h;
typedef __attribute__((ext_vector_type(8)))  float    v8f;

#define ENT_TYPES 9
#define INNER     64
#define HID       1024
#define NOUT      1152   // 9 * 64 * 2
#define SEQ       512
#define BATCH     16
#define MTOT      8192   // BATCH * SEQ
#define BIG_NEG   1000000000000.0f

union F16Frag { v16h v; int4 q[2]; };           // 8 VGPRs, two b128 halves
union H4     { _Float16 h[4]; int2 i2; };       // 4 halves = one 8B LDS store
union H2     { _Float16 h[2]; int  i;  };       // rope pair = one 4B LDS store

// ---------------------------------------------------------------------------
// Kernel 1: Y = X @ W + b  (8192x1024 @ 1024x1152, fp16 WMMA, fp32 accum)
// Block 256 thr (8 waves). Tile 128Mx128N, K staged by 32 with register
// double-buffering. Wave (wm,wn) = 32x64 patch = 2x4 v_wmma_f32_16x16x32_f16.
// f32->f16 conversion fused into the LDS staging; W stored transposed [n][k]
// so B-frags are two contiguous b128 LDS loads.
// ---------------------------------------------------------------------------
#define XS_STR 40   // 32 K-halves + 8 pad (16B-aligned rows, spread banks)
#define WS_STR 40

__global__ __launch_bounds__(256) void gp_proj_gemm(
    const float* __restrict__ X, const float* __restrict__ W,
    const float* __restrict__ bias, float* __restrict__ Y)
{
  __shared__ __align__(16) _Float16 Xs [128 * XS_STR];   // [m][k]
  __shared__ __align__(16) _Float16 WsT[128 * WS_STR];   // [n][k]

  const int tid  = threadIdx.x;
  const int lane = tid & 31;
  const int wave = tid >> 5;
  const int l16  = lane & 15;
  const int lhi  = lane >> 4;

  const int n0 = blockIdx.x * 128;
  const int m0 = blockIdx.y * 128;
  const int wm = wave >> 1;            // 0..3 : 32-row strip
  const int wn = wave & 1;             // 0..1 : 64-col strip

  v8f zero = {};
  v8f acc[2][4];
  #pragma unroll
  for (int i = 0; i < 2; ++i)
    #pragma unroll
    for (int j = 0; j < 4; ++j) acc[i][j] = zero;

  float4 xr[4], wr[4];
  auto loadTiles = [&](int k0) {
    #pragma unroll
    for (int it = 0; it < 4; ++it) {           // X tile 128x32 f32
      int idx = tid + it * 256;
      int row = idx >> 3, c4 = (idx & 7) << 2;
      xr[it] = *(const float4*)(X + (size_t)(m0 + row) * HID + k0 + c4);
    }
    #pragma unroll
    for (int it = 0; it < 4; ++it) {           // W tile 32x128 f32
      int idx = tid + it * 256;
      int kr = idx >> 5, c4 = (idx & 31) << 2;
      wr[it] = *(const float4*)(W + (size_t)(k0 + kr) * NOUT + n0 + c4);
    }
  };

  loadTiles(0);

  for (int step = 0; step < HID / 32; ++step) {
    // ---- store staged registers to LDS with f32->f16 convert ----
    #pragma unroll
    for (int it = 0; it < 4; ++it) {
      int idx = tid + it * 256;
      int row = idx >> 3, c4 = (idx & 7) << 2;
      H4 u;
      u.h[0] = (_Float16)xr[it].x; u.h[1] = (_Float16)xr[it].y;
      u.h[2] = (_Float16)xr[it].z; u.h[3] = (_Float16)xr[it].w;
      *(int2*)&Xs[row * XS_STR + c4] = u.i2;
    }
    #pragma unroll
    for (int it = 0; it < 4; ++it) {           // transpose W into [n][k]
      int idx = tid + it * 256;
      int kr = idx >> 5, c4 = (idx & 31) << 2;
      WsT[(c4 + 0) * WS_STR + kr] = (_Float16)wr[it].x;
      WsT[(c4 + 1) * WS_STR + kr] = (_Float16)wr[it].y;
      WsT[(c4 + 2) * WS_STR + kr] = (_Float16)wr[it].z;
      WsT[(c4 + 3) * WS_STR + kr] = (_Float16)wr[it].w;
    }
    __syncthreads();

    if (step + 1 < HID / 32) loadTiles((step + 1) * 32);   // prefetch next K

    // ---- fragments + 2x4 WMMA ----
    F16Frag a[2], bf[4];
    #pragma unroll
    for (int i = 0; i < 2; ++i) {
      int row = wm * 32 + i * 16 + l16;
      a[i].q[0] = *(const int4*)&Xs[row * XS_STR + lhi * 8];        // K 0-7 / 8-15
      a[i].q[1] = *(const int4*)&Xs[row * XS_STR + 16 + lhi * 8];   // K 16-23 / 24-31
    }
    #pragma unroll
    for (int j = 0; j < 4; ++j) {
      int n = wn * 64 + j * 16 + l16;
      bf[j].q[0] = *(const int4*)&WsT[n * WS_STR + lhi * 8];        // K 0-7 / 8-15
      bf[j].q[1] = *(const int4*)&WsT[n * WS_STR + 16 + lhi * 8];   // K 16-23 / 24-31
    }
    #pragma unroll
    for (int i = 0; i < 2; ++i)
      #pragma unroll
      for (int j = 0; j < 4; ++j)
        acc[i][j] = __builtin_amdgcn_wmma_f32_16x16x32_f16(
            false, a[i].v, false, bf[j].v, (short)0, acc[i][j], false, false);
    __syncthreads();
  }

  // ---- epilogue: + bias, store Y (fp32) ----
  #pragma unroll
  for (int j = 0; j < 4; ++j) {
    const int col = n0 + wn * 64 + j * 16 + l16;
    const float bv = bias[col];
    #pragma unroll
    for (int i = 0; i < 2; ++i) {
      #pragma unroll
      for (int r = 0; r < 8; ++r) {
        const int row = m0 + wm * 32 + i * 16 + r + lhi * 8;
        Y[(size_t)row * NOUT + col] = acc[i][j][r] + bv;
      }
    }
  }
}

// ---------------------------------------------------------------------------
// Kernel 2: per (b,h): logits = rope(q) @ rope(k)^T, masks, /8
// RoPE applied in fp32 during global->LDS staging (pairs are contiguous
// float2), converted to f16. Wave (wm,wn) = 64x32 patch = 4x2 WMMAs x 2
// K-steps of 32. Accumulate fp32, mask+scale in epilogue.
// ---------------------------------------------------------------------------
#define QS_STR 72   // 64 K-halves + 8 pad

__global__ __launch_bounds__(256) void gp_attn(
    const float* __restrict__ Y, const float* __restrict__ pm,
    float* __restrict__ out)
{
  __shared__ __align__(16) _Float16 Qs[128 * QS_STR];
  __shared__ __align__(16) _Float16 Ks[128 * QS_STR];

  const int tid  = threadIdx.x;
  const int lane = tid & 31;
  const int wave = tid >> 5;
  const int l16  = lane & 15;
  const int lhi  = lane >> 4;

  const int tm = (blockIdx.x >> 2) * 128;   // query rows
  const int tn = (blockIdx.x & 3) * 128;    // key rows
  const int h  = blockIdx.y;
  const int b  = blockIdx.z;

  const float LN10000 = 9.210340371976184f;

  // ---- load + RoPE + f16 convert: 128 rows x 32 pairs for q and k ----
  for (int idx = tid; idx < 128 * 32; idx += 256) {
    const int row = idx >> 5;     // 0..127
    const int pi  = idx & 31;     // pair index 0..31
    const float inv_freq = expf(-(float)pi * (1.0f / 32.0f) * LN10000);
    {
      const int s = tm + row;
      float c, sn; sincosf((float)s * inv_freq, &sn, &c);
      const float2 v = *(const float2*)(Y + (size_t)(b * SEQ + s) * NOUT + h * 128 + 2 * pi);
      H2 u;
      u.h[0] = (_Float16)(v.x * c - v.y * sn);
      u.h[1] = (_Float16)(v.y * c + v.x * sn);
      *(int*)&Qs[row * QS_STR + 2 * pi] = u.i;
    }
    {
      const int s = tn + row;
      float c, sn; sincosf((float)s * inv_freq, &sn, &c);
      const float2 v = *(const float2*)(Y + (size_t)(b * SEQ + s) * NOUT + h * 128 + INNER + 2 * pi);
      H2 u;
      u.h[0] = (_Float16)(v.x * c - v.y * sn);
      u.h[1] = (_Float16)(v.y * c + v.x * sn);
      *(int*)&Ks[row * QS_STR + 2 * pi] = u.i;
    }
  }
  __syncthreads();

  const int wm = wave >> 2;   // 0..1 : 64-row strip
  const int wn = wave & 3;    // 0..3 : 32-col strip

  v8f zero = {};
  v8f acc[4][2];
  #pragma unroll
  for (int i = 0; i < 4; ++i)
    #pragma unroll
    for (int j = 0; j < 2; ++j) acc[i][j] = zero;

  #pragma unroll
  for (int kk = 0; kk < INNER; kk += 32) {
    F16Frag a[4], bf[2];
    #pragma unroll
    for (int i = 0; i < 4; ++i) {
      int row = wm * 64 + i * 16 + l16;
      a[i].q[0] = *(const int4*)&Qs[row * QS_STR + kk + lhi * 8];
      a[i].q[1] = *(const int4*)&Qs[row * QS_STR + kk + 16 + lhi * 8];
    }
    #pragma unroll
    for (int j = 0; j < 2; ++j) {
      int n = wn * 32 + j * 16 + l16;     // B(k,n) = Kr[n][k]
      bf[j].q[0] = *(const int4*)&Ks[n * QS_STR + kk + lhi * 8];
      bf[j].q[1] = *(const int4*)&Ks[n * QS_STR + kk + 16 + lhi * 8];
    }
    #pragma unroll
    for (int i = 0; i < 4; ++i)
      #pragma unroll
      for (int j = 0; j < 2; ++j)
        acc[i][j] = __builtin_amdgcn_wmma_f32_16x16x32_f16(
            false, a[i].v, false, bf[j].v, (short)0, acc[i][j], false, false);
  }

  // ---- epilogue: padding mask, causal (n<m), scale 1/8, store ----
  float* obase = out + ((size_t)(b * ENT_TYPES + h)) * SEQ * SEQ;
  #pragma unroll
  for (int j = 0; j < 2; ++j) {
    const int n   = tn + wn * 32 + j * 16 + l16;
    const float p = pm[b * SEQ + n];
    const float neg = (1.0f - p) * BIG_NEG;
    #pragma unroll
    for (int i = 0; i < 4; ++i) {
      #pragma unroll
      for (int r = 0; r < 8; ++r) {
        const int m = tm + wm * 64 + i * 16 + r + lhi * 8;
        float v = acc[i][j][r] * p - neg;
        if (n < m) v -= BIG_NEG;
        obase[(size_t)m * SEQ + n] = v * 0.125f;
      }
    }
  }
}

// ---------------------------------------------------------------------------
extern "C" void kernel_launch(void* const* d_in, const int* in_sizes, int n_in,
                              void* d_out, int out_size, void* d_ws, size_t ws_size,
                              hipStream_t stream) {
  const float* X     = (const float*)d_in[0];  // (16,512,1024)
  const float* pmask = (const float*)d_in[1];  // (16,512)
  const float* W     = (const float*)d_in[2];  // (1024,1152)
  const float* bias  = (const float*)d_in[3];  // (1152,)
  float* out = (float*)d_out;                  // (16,9,512,512)
  float* Y   = (float*)d_ws;                   // (8192,1152) fp32 = 37.75 MB

  gp_proj_gemm<<<dim3(NOUT / 128, MTOT / 128), 256, 0, stream>>>(X, W, bias, Y);
  gp_attn<<<dim3(16, ENT_TYPES, BATCH), 256, 0, stream>>>(Y, pmask, out);
}
